// MultiHeadAttention_86208583565316
// MI455X (gfx1250) — matrix operand order repaired and checked by reading, not compiled, not run on hardware
//
#include <hip/hip_runtime.h>
#include <hip/hip_bf16.h>

// Problem constants (match reference)
#define BB     2
#define LL     2048
#define HH     1024
#define NHEAD  16
#define KVHEAD 4
#define HD     64
#define GQ     4          // NHEAD / KVHEAD

typedef __attribute__((ext_vector_type(16))) __bf16 bf16x16;
typedef __attribute__((ext_vector_type(8)))  float  f32x8;

union Frag { bf16x16 v; uint4 q[2]; };

// ---------------------------------------------------------------------------
// CDNA5 async global->LDS staging (ASYNCcnt path), guarded with fallback.
// Builtin signature (from clang diagnostics): (v4i AS1*, v4i AS3*, Ii, Ii)
// where v4i = int __attribute__((vector_size(16))).
// ---------------------------------------------------------------------------
#if defined(__has_builtin)
# if __has_builtin(__builtin_amdgcn_global_load_async_to_lds_b128)
#  define HAVE_ASYNC_LDS 1
# endif
#endif
#ifndef HAVE_ASYNC_LDS
# define HAVE_ASYNC_LDS 0
#endif

typedef int v4i_ __attribute__((vector_size(16)));
#define AS1P(p) ((__attribute__((address_space(1))) v4i_*)(v4i_*)(void*)(p))
#define AS3P(p) ((__attribute__((address_space(3))) v4i_*)(v4i_*)(void*)(p))

__device__ __forceinline__ void wait_async() {
#if HAVE_ASYNC_LDS
# if __has_builtin(__builtin_amdgcn_s_wait_asynccnt)
    __builtin_amdgcn_s_wait_asynccnt(0);
# else
    asm volatile("s_wait_asynccnt 0x0" ::: "memory");
# endif
#endif
}

// ---------------------------------------------------------------------------
// fp32 -> bf16 conversion (vectorized x4)
// ---------------------------------------------------------------------------
__global__ void cvt_f32_bf16(const float* __restrict__ in, __bf16* __restrict__ out, int n) {
    int i = (blockIdx.x * blockDim.x + threadIdx.x) * 4;
    if (i < n) {
        float4 f = *(const float4*)(in + i);
        out[i + 0] = (__bf16)f.x;
        out[i + 1] = (__bf16)f.y;
        out[i + 2] = (__bf16)f.z;
        out[i + 3] = (__bf16)f.w;
    }
}

// ---------------------------------------------------------------------------
// Tiled bf16 WMMA GEMM:  C[M,N] = A[M,K] @ Bw[N,K]^T + bias[N]
//   mode 0: write bf16, head-major layout [b, head, l, hd]   (QKV projections)
//   mode 1: write fp32, row-major [M,N]                      (output projection)
// Tile: 128(M) x 64(N), BK=32, double-buffered LDS.
// 4 waves (128 thr); each wave: 32(M) x 64(N) = 2x4 WMMA tiles -> 8 wmma/step.
// ---------------------------------------------------------------------------
__global__ __launch_bounds__(128)
void gemm_bf16(const __bf16* __restrict__ A, const __bf16* __restrict__ Bw,
               const float* __restrict__ bias, __bf16* __restrict__ outBf,
               float* __restrict__ outF, int M, int N, int K, int mode) {
    const int tid  = threadIdx.x;
    const int w    = tid >> 5;
    const int lane = tid & 31;
    const int hlf  = (lane >> 4) & 1;
    const int l16  = lane & 15;

    const int rowBase = blockIdx.y * 128;
    const int colBase = blockIdx.x * 64;

    __shared__ __bf16 As[2][128][40];   // 32 + 8 pad (row stride 80B)
    __shared__ __bf16 Bs[2][64][40];

    f32x8 acc[2][4] = {};

    // staging coordinates: A = 1 row/thread (32 elems), B = half row/thread
    const int ar = tid;
    const int br = tid >> 1, bc = (tid & 1) * 16;
    const __bf16* Ag = A  + (size_t)(rowBase + ar) * K;
    const __bf16* Bg = Bw + (size_t)(colBase + br) * K + bc;

    auto stage = [&](int kk, __bf16 (*Asb)[40], __bf16 (*Bsb)[40]) {
#if HAVE_ASYNC_LDS
#pragma unroll
        for (int c = 0; c < 4; ++c)
            __builtin_amdgcn_global_load_async_to_lds_b128(
                AS1P(Ag + kk + c * 8), AS3P(&Asb[ar][c * 8]), 0, 0);
        __builtin_amdgcn_global_load_async_to_lds_b128(
            AS1P(Bg + kk), AS3P(&Bsb[br][bc]), 0, 0);
        __builtin_amdgcn_global_load_async_to_lds_b128(
            AS1P(Bg + kk + 8), AS3P(&Bsb[br][bc + 8]), 0, 0);
#else
        uint4 a0 = *(const uint4*)(Ag + kk);
        uint4 a1 = *(const uint4*)(Ag + kk + 8);
        uint4 a2 = *(const uint4*)(Ag + kk + 16);
        uint4 a3 = *(const uint4*)(Ag + kk + 24);
        uint4 b0 = *(const uint4*)(Bg + kk);
        uint4 b1 = *(const uint4*)(Bg + kk + 8);
        *(uint4*)&Asb[ar][0]       = a0;
        *(uint4*)&Asb[ar][8]       = a1;
        *(uint4*)&Asb[ar][16]      = a2;
        *(uint4*)&Asb[ar][24]      = a3;
        *(uint4*)&Bsb[br][bc]      = b0;
        *(uint4*)&Bsb[br][bc + 8]  = b1;
#endif
    };

    stage(0, As[0], Bs[0]);
    wait_async();
    __syncthreads();

    for (int k0 = 0; k0 < K; k0 += 32) {
        const int cur = (k0 >> 5) & 1;
        if (k0 + 32 < K) {
            stage(k0 + 32, As[cur ^ 1], Bs[cur ^ 1]);
            __builtin_prefetch(Ag + k0 + 96, 0, 3);   // near-scope prefetch 2 tiles ahead
            __builtin_prefetch(Bg + k0 + 96, 0, 3);
        }

        Frag a[2], bf[4];
#pragma unroll
        for (int i = 0; i < 2; ++i) {
            // A fragment (16x32): lane half h -> K chunks {h*8.., 16+h*8..}
            const __bf16* pa = &As[cur][w * 32 + i * 16 + l16][hlf * 8];
            a[i].q[0] = *(const uint4*)pa;
            a[i].q[1] = *(const uint4*)(pa + 16);
        }
#pragma unroll
        for (int j = 0; j < 4; ++j) {
            // B fragment (32x16): lane half h -> K chunks {h*8.., 16+h*8..}
            const __bf16* pb = &Bs[cur][j * 16 + l16][hlf * 8];
            bf[j].q[0] = *(const uint4*)pb;
            bf[j].q[1] = *(const uint4*)(pb + 16);
        }
#pragma unroll
        for (int i = 0; i < 2; ++i)
#pragma unroll
            for (int j = 0; j < 4; ++j)
                acc[i][j] = __builtin_amdgcn_wmma_f32_16x16x32_bf16(
                    false, a[i].v, false, bf[j].v, (short)0, acc[i][j], false, false);

        wait_async();
        __syncthreads();
    }

    const int heads = N >> 6;
#pragma unroll
    for (int i = 0; i < 2; ++i)
#pragma unroll
        for (int j = 0; j < 4; ++j)
#pragma unroll
            for (int t = 0; t < 8; ++t) {
                int m = rowBase + w * 32 + i * 16 + hlf * 8 + t;
                int n = colBase + j * 16 + l16;
                float v = acc[i][j][t] + bias[n];
                if (mode == 0) {
                    int b = m >> 11, l = m & (LL - 1);
                    int head = n >> 6, hd = n & 63;
                    outBf[(((size_t)b * heads + head) * LL + l) * HD + hd] = (__bf16)v;
                } else {
                    outF[(size_t)m * N + n] = v;
                }
            }
}

// ---------------------------------------------------------------------------
// Flash-attention (GQA): per block = one (b, head) x 64 query rows.
// 4 waves x 16 query rows; key blocks of 64; online softmax; WMMA for QK^T + PV.
// ---------------------------------------------------------------------------
__global__ __launch_bounds__(128)
void attn_kernel(const __bf16* __restrict__ Q, const __bf16* __restrict__ Kg,
                 const __bf16* __restrict__ Vg, const float* __restrict__ mask,
                 __bf16* __restrict__ ctx) {
    const int qb  = blockIdx.x;              // query block 0..31
    const int bh  = blockIdx.y;              // b*NHEAD + nh
    const int b   = bh / NHEAD;
    const int nh  = bh % NHEAD;
    const int kv  = nh / GQ;
    const int tid = threadIdx.x;
    const int w   = tid >> 5;
    const int lane = tid & 31;
    const int hlf  = (lane >> 4) & 1;
    const int l16  = lane & 15;

    __shared__ __bf16 Ks[64][HD + 8];        // key rows x head dim (QK^T B-frags)
    __shared__ __bf16 Vs[HD][64 + 8];        // transposed V (PV B-frags)
    __shared__ __bf16 Ps[4][16][64 + 8];     // per-wave P staging (C->A layout)

    // Q A-fragments for this wave's 16 rows, K split in two 32-chunks of HD
    const int qRow0 = qb * 64 + w * 16;
    const __bf16* qrow = Q + ((size_t)bh * LL + qRow0 + l16) * HD;
    Frag qa[2];
#pragma unroll
    for (int s = 0; s < 2; ++s) {
        qa[s].q[0] = *(const uint4*)(qrow + s * 32 + hlf * 8);
        qa[s].q[1] = *(const uint4*)(qrow + s * 32 + 16 + hlf * 8);
    }

    float mrow[8], lrow[8];
    f32x8 o[4] = {};
#pragma unroll
    for (int t = 0; t < 8; ++t) { mrow[t] = -3.0e38f; lrow[t] = 0.f; }

    const __bf16* Kbase = Kg + (size_t)(b * KVHEAD + kv) * LL * HD;
    const __bf16* Vbase = Vg + (size_t)(b * KVHEAD + kv) * LL * HD;

    const int r  = tid >> 1;
    const int c0 = (tid & 1) * 32;

    for (int kb = 0; kb < LL / 64; ++kb) {
        const int k0 = kb * 64;
        __syncthreads();
        {   // K tile: async DMA to LDS if available; V tile: transposed scatter
#if HAVE_ASYNC_LDS
#pragma unroll
            for (int c = 0; c < 4; ++c)
                __builtin_amdgcn_global_load_async_to_lds_b128(
                    AS1P(Kbase + (size_t)(k0 + r) * HD + c0 + c * 8),
                    AS3P(&Ks[r][c0 + c * 8]), 0, 0);
#else
            const uint4* ksrc = (const uint4*)(Kbase + (size_t)(k0 + r) * HD + c0);
            uint4* kdst = (uint4*)&Ks[r][c0];
            kdst[0] = ksrc[0]; kdst[1] = ksrc[1]; kdst[2] = ksrc[2]; kdst[3] = ksrc[3];
#endif
            union { uint4 q[4]; __bf16 h[32]; } vt;
            const uint4* vsrc = (const uint4*)(Vbase + (size_t)(k0 + r) * HD + c0);
            vt.q[0] = vsrc[0]; vt.q[1] = vsrc[1]; vt.q[2] = vsrc[2]; vt.q[3] = vsrc[3];
#pragma unroll
            for (int i = 0; i < 32; ++i) Vs[c0 + i][r] = vt.h[i];
        }
        wait_async();
        __syncthreads();

        // S = Q K^T * (1/8) + mask   (4 n-tiles of 16 keys, 2 k-steps of 32)
        float sv[4][8];
#pragma unroll
        for (int nt = 0; nt < 4; ++nt) {
            f32x8 sacc = {};
#pragma unroll
            for (int s = 0; s < 2; ++s) {
                Frag bk;
                const __bf16* p = &Ks[nt * 16 + l16][s * 32 + hlf * 8];
                bk.q[0] = *(const uint4*)p;
                bk.q[1] = *(const uint4*)(p + 16);
                sacc = __builtin_amdgcn_wmma_f32_16x16x32_bf16(
                    false, qa[s].v, false, bk.v, (short)0, sacc, false, false);
            }
#pragma unroll
            for (int t = 0; t < 8; ++t) {
                int qr = qRow0 + hlf * 8 + t;
                int kc = k0 + nt * 16 + l16;
                float mv = mask[((size_t)b * LL + qr) * LL + kc];
                sv[nt][t] = sacc[t] * 0.125f + mv;
            }
        }

        // online softmax update (reductions stay within 16-lane half: xor 1,2,4,8)
        float corr[8];
#pragma unroll
        for (int t = 0; t < 8; ++t) {
            float mx = fmaxf(fmaxf(sv[0][t], sv[1][t]), fmaxf(sv[2][t], sv[3][t]));
            mx = fmaxf(mx, __shfl_xor(mx, 1, 32));
            mx = fmaxf(mx, __shfl_xor(mx, 2, 32));
            mx = fmaxf(mx, __shfl_xor(mx, 4, 32));
            mx = fmaxf(mx, __shfl_xor(mx, 8, 32));
            float mnew = fmaxf(mrow[t], mx);
            corr[t] = expf(mrow[t] - mnew);
            mrow[t] = mnew;
            float rs = 0.f;
#pragma unroll
            for (int nt = 0; nt < 4; ++nt) {
                float pe = expf(sv[nt][t] - mnew);
                sv[nt][t] = pe;
                rs += pe;
            }
            rs += __shfl_xor(rs, 1, 32);
            rs += __shfl_xor(rs, 2, 32);
            rs += __shfl_xor(rs, 4, 32);
            rs += __shfl_xor(rs, 8, 32);
            lrow[t] = lrow[t] * corr[t] + rs;
        }

        // rescale O, stage P through LDS (C layout -> A layout); Ps is per-wave
        // private and LDS ops are in-order within a wave, so a wave-level
        // scheduling barrier suffices (no workgroup barrier needed here).
#pragma unroll
        for (int nt = 0; nt < 4; ++nt)
#pragma unroll
            for (int t = 0; t < 8; ++t) {
                o[nt][t] *= corr[t];
                Ps[w][hlf * 8 + t][nt * 16 + l16] = (__bf16)sv[nt][t];
            }
        __builtin_amdgcn_wave_barrier();

        // O += P @ V   (P: 16x64 over keys, V^T in LDS)
        Frag pa[2];
        const __bf16* prow = &Ps[w][l16][0];
#pragma unroll
        for (int s = 0; s < 2; ++s) {
            pa[s].q[0] = *(const uint4*)(prow + s * 32 + hlf * 8);
            pa[s].q[1] = *(const uint4*)(prow + s * 32 + 16 + hlf * 8);
        }
#pragma unroll
        for (int nt = 0; nt < 4; ++nt)
#pragma unroll
            for (int s = 0; s < 2; ++s) {
                Frag vb;
                const __bf16* p = &Vs[nt * 16 + l16][s * 32 + hlf * 8];
                vb.q[0] = *(const uint4*)p;
                vb.q[1] = *(const uint4*)(p + 16);
                o[nt] = __builtin_amdgcn_wmma_f32_16x16x32_bf16(
                    false, pa[s].v, false, vb.v, (short)0, o[nt], false, false);
            }
    }

    // normalize and write ctx in token-major [b, l, nh*64+hd] bf16
#pragma unroll
    for (int nt = 0; nt < 4; ++nt)
#pragma unroll
        for (int t = 0; t < 8; ++t) {
            float v = o[nt][t] / lrow[t];
            int l = qRow0 + hlf * 8 + t;
            int h = nh * HD + nt * 16 + l16;
            ctx[((size_t)b * LL + l) * HH + h] = (__bf16)v;
        }
}

// ---------------------------------------------------------------------------
extern "C" void kernel_launch(void* const* d_in, const int* in_sizes, int n_in,
                              void* d_out, int out_size, void* d_ws, size_t ws_size,
                              hipStream_t stream) {
    const float* x    = (const float*)d_in[0];
    const float* mask = (const float*)d_in[1];
    const float* Wq   = (const float*)d_in[2];
    const float* bq   = (const float*)d_in[3];
    const float* Wk   = (const float*)d_in[4];
    const float* bk   = (const float*)d_in[5];
    const float* Wv   = (const float*)d_in[6];
    const float* bv   = (const float*)d_in[7];
    const float* Wo   = (const float*)d_in[8];
    const float* bo   = (const float*)d_in[9];
    float* out = (float*)d_out;

    char* ws = (char*)d_ws;
    size_t off = 0;
    auto alloc = [&](size_t bytes) -> void* {
        void* p = ws + off;
        off += (bytes + 255) & ~(size_t)255;
        return p;
    };
    const int M = BB * LL;                       // 4096 tokens
    __bf16* Xb   = (__bf16*)alloc((size_t)M * HH * 2);
    __bf16* Wqb  = (__bf16*)alloc((size_t)HH * HH * 2);
    __bf16* Wkb  = (__bf16*)alloc((size_t)KVHEAD * HD * HH * 2);
    __bf16* Wvb  = (__bf16*)alloc((size_t)KVHEAD * HD * HH * 2);
    __bf16* Wob  = (__bf16*)alloc((size_t)HH * HH * 2);
    __bf16* Qb   = (__bf16*)alloc((size_t)BB * NHEAD  * LL * HD * 2);
    __bf16* Kb   = (__bf16*)alloc((size_t)BB * KVHEAD * LL * HD * 2);
    __bf16* Vb   = (__bf16*)alloc((size_t)BB * KVHEAD * LL * HD * 2);
    __bf16* Ctxb = (__bf16*)alloc((size_t)M * HH * 2);

    // fp32 -> bf16 conversions
    struct { const float* src; __bf16* dst; int n; } cv[5] = {
        { x,  Xb,  M * HH },
        { Wq, Wqb, HH * HH },
        { Wk, Wkb, KVHEAD * HD * HH },
        { Wv, Wvb, KVHEAD * HD * HH },
        { Wo, Wob, HH * HH },
    };
    for (int i = 0; i < 5; ++i) {
        int thr = cv[i].n / 4;
        cvt_f32_bf16<<<(thr + 255) / 256, 256, 0, stream>>>(cv[i].src, cv[i].dst, cv[i].n);
    }

    // Q/K/V projections (bf16 WMMA, fp32 accumulate), head-major bf16 outputs
    dim3 blk(128);
    gemm_bf16<<<dim3(HH / 64, M / 128), blk, 0, stream>>>(Xb, Wqb, bq, Qb, nullptr, M, HH, HH, 0);
    gemm_bf16<<<dim3((KVHEAD * HD) / 64, M / 128), blk, 0, stream>>>(Xb, Wkb, bk, Kb, nullptr, M, KVHEAD * HD, HH, 0);
    gemm_bf16<<<dim3((KVHEAD * HD) / 64, M / 128), blk, 0, stream>>>(Xb, Wvb, bv, Vb, nullptr, M, KVHEAD * HD, HH, 0);

    // flash attention
    attn_kernel<<<dim3(LL / 64, BB * NHEAD), dim3(128), 0, stream>>>(Qb, Kb, Vb, mask, Ctxb);

    // output projection -> fp32 d_out
    gemm_bf16<<<dim3(HH / 64, M / 128), blk, 0, stream>>>(Ctxb, Wob, bo, nullptr, out, M, HH, HH, 1);
}
